// SemanticAnswerDistillation_67594195304459
// MI455X (gfx1250) — matrix operand order repaired and checked by reading, not compiled, not run on hardware
//
#include <hip/hip_runtime.h>
#include <hip/hip_bf16.h>
#include <math.h>

#define DIM     768
#define BATCH   1024
#define NANS    128000
#define KTOP    10
#define NCHUNKS 64
#define CHUNK   2000   /* NANS / NCHUNKS */
#define MT      64     /* BATCH / 16     */

typedef __attribute__((ext_vector_type(16))) __bf16         v16bf;
typedef __attribute__((ext_vector_type(8)))  float          v8f;
typedef __attribute__((ext_vector_type(4)))  unsigned       u32x4;
typedef __attribute__((ext_vector_type(4)))  int            i32x4;
typedef __attribute__((ext_vector_type(4)))  unsigned short u16x4;
typedef __attribute__((ext_vector_type(4)))  float          f32x4;

union Frag { v16bf v; u32x4 q[2]; };

// Native hardware bf16 convert (v_cvt_*bf16_f32 on gfx1250), RNE.
__device__ __forceinline__ __bf16 f2bf(float f) { return (__bf16)f; }
__device__ __forceinline__ unsigned short f2bfu(float f) {
  return __builtin_bit_cast(unsigned short, (__bf16)f);
}

__device__ __forceinline__ v8f wmma_bf16(v16bf a, v16bf b, v8f c) {
  return __builtin_amdgcn_wmma_f32_16x16x32_bf16(false, a, false, b, (short)0, c, false, false);
}

#if defined(__has_builtin)
#if __has_builtin(__builtin_amdgcn_global_load_async_to_lds_b128) && \
    __has_builtin(__builtin_amdgcn_s_wait_asynccnt)
#define USE_ASYNC_LDS 1
#endif
#endif

// ---------------------------------------------------------------------------
// C[M,N] = A[M,K] @ B[N,K]^T (+bias) (+C if accum), bf16 WMMA, f32 accum.
// One wave per 16x16 output tile. grid = (N/16, M/16), block = 32.
// ---------------------------------------------------------------------------
__global__ void gemm_bt_kernel(const float* __restrict__ A, int lda,
                               const float* __restrict__ B, int ldb,
                               const float* __restrict__ bias,
                               float* __restrict__ C, int N, int K, int accum)
{
  int lane = threadIdx.x & 31;
  int n0 = blockIdx.x * 16;
  int m0 = blockIdx.y * 16;
  int r   = lane & 15;            // A row / C col / B col within tile
  int kbA = (lane >> 4) * 16;     // A K split: contiguous, matches B
  int kbB = (lane >> 4) * 16;     // B K split: contiguous, same as A
  int rb  = (lane >> 4) * 8;      // C row base per ISA 32-bit C layout

  v8f c = {};
  if (accum) {
#pragma unroll
    for (int j = 0; j < 8; ++j) c[j] = C[(size_t)(m0 + rb + j) * N + n0 + r];
  }
  const float* arow = A + (size_t)(m0 + r) * lda;
  const float* brow = B + (size_t)(n0 + r) * ldb;
  for (int k0 = 0; k0 < K; k0 += 32) {
    v16bf a, b;
#pragma unroll
    for (int i = 0; i < 8; ++i) {
      a[i]     = f2bf(arow[k0 + kbA + i]);
      a[8 + i] = f2bf(arow[k0 + kbA + 8 + i]);
      b[i]     = f2bf(brow[k0 + kbB + i]);
      b[8 + i] = f2bf(brow[k0 + kbB + 8 + i]);
    }
    c = wmma_bf16(a, b, c);
  }
  float bb = bias ? bias[n0 + r] : 0.0f;
#pragma unroll
  for (int j = 0; j < 8; ++j)
    C[(size_t)(m0 + rb + j) * N + n0 + r] = c[j] + bb;
}

// ---------------------------------------------------------------------------
// Row-wise L2 normalize (clamp_min eps) f32 -> bf16 (stored as ushort).
// One wave32 per row, float4 loads / bf16x4 stores. block = 256 (8 rows).
// ---------------------------------------------------------------------------
__global__ void l2norm_bf16_kernel(const float* __restrict__ X,
                                   unsigned short* __restrict__ Y, int rows)
{
  int wave = threadIdx.x >> 5, lane = threadIdx.x & 31;
  int row = blockIdx.x * 8 + wave;
  if (row >= rows) return;
  const float* x = X + (size_t)row * DIM;
  f32x4 vals[6];
  float ss = 0.f;
#pragma unroll
  for (int i = 0; i < 6; ++i) {
    vals[i] = *(const f32x4*)&x[lane * 4 + 128 * i];
    ss += vals[i].x * vals[i].x + vals[i].y * vals[i].y +
          vals[i].z * vals[i].z + vals[i].w * vals[i].w;
  }
#pragma unroll
  for (int off = 16; off > 0; off >>= 1) ss += __shfl_xor(ss, off, 32);
  float inv = 1.0f / fmaxf(sqrtf(ss), 1e-12f);
  unsigned short* y = Y + (size_t)row * DIM;
#pragma unroll
  for (int i = 0; i < 6; ++i) {
    u16x4 o;
    o[0] = f2bfu(vals[i].x * inv);
    o[1] = f2bfu(vals[i].y * inv);
    o[2] = f2bfu(vals[i].z * inv);
    o[3] = f2bfu(vals[i].w * inv);
    *(u16x4*)&y[lane * 4 + 128 * i] = o;
  }
}

// ---------------------------------------------------------------------------
// LayerNorm (eps 1e-5, affine) + exact GELU. One wave per row.
// ---------------------------------------------------------------------------
__global__ void ln_gelu_kernel(const float* __restrict__ H,
                               const float* __restrict__ g, const float* __restrict__ b,
                               float* __restrict__ Out)
{
  int wave = threadIdx.x >> 5, lane = threadIdx.x & 31;
  int row = blockIdx.x * 8 + wave;
  if (row >= BATCH) return;
  const float* x = H + (size_t)row * DIM;
  float vals[24];
  float s = 0.f;
#pragma unroll
  for (int i = 0; i < 24; ++i) { vals[i] = x[lane + 32 * i]; s += vals[i]; }
#pragma unroll
  for (int off = 16; off > 0; off >>= 1) s += __shfl_xor(s, off, 32);
  float mu = s * (1.0f / (float)DIM);
  float vs = 0.f;
#pragma unroll
  for (int i = 0; i < 24; ++i) { float d = vals[i] - mu; vs += d * d; }
#pragma unroll
  for (int off = 16; off > 0; off >>= 1) vs += __shfl_xor(vs, off, 32);
  float rstd = rsqrtf(vs * (1.0f / (float)DIM) + 1e-5f);
  float* o = Out + (size_t)row * DIM;
#pragma unroll
  for (int i = 0; i < 24; ++i) {
    int cidx = lane + 32 * i;
    float yv = (vals[i] - mu) * rstd * g[cidx] + b[cidx];
    o[cidx] = 0.5f * yv * (1.0f + erff(yv * 0.70710678118654752f));
  }
}

// ---------------------------------------------------------------------------
// Fused similarity GEMM + mask + per-chunk top-10.
// grid = (MT mtiles, NCHUNKS chunks), block = 256 (8 waves).
// Each wave: 16x32 output (two 16x16 bf16 WMMA tiles sharing the A fragment)
// over K=768. Per-row top-10 kept in registers of threads 0..15, scanned from
// an LDS score staging buffer. Query tile loaded via async global->LDS.
// ---------------------------------------------------------------------------
__global__ void __launch_bounds__(256) sim_topk_kernel(
    const unsigned short* __restrict__ Qn, const unsigned short* __restrict__ Bn,
    const float* __restrict__ mask, float* __restrict__ partS, int* __restrict__ partI)
{
  __shared__ __align__(16) unsigned short As[16 * DIM];   // 24 KB query tile (bf16)
  __shared__ float Sc[8 * 512];                            // 16 KB score staging
  int tid = threadIdx.x, lane = tid & 31, wave = tid >> 5;
  int mtile = blockIdx.x, chunk = blockIdx.y;
  int m0 = mtile * 16;

#ifdef USE_ASYNC_LDS
  {
    const unsigned short* gsrc = Qn + (size_t)m0 * DIM;
    for (int i = tid; i < (16 * DIM) / 8; i += 256) {
      __builtin_amdgcn_global_load_async_to_lds_b128(
          (__attribute__((address_space(1))) i32x4*)(gsrc + i * 8),
          (__attribute__((address_space(3))) i32x4*)(As + i * 8), 0, 0);
    }
    __builtin_amdgcn_s_wait_asynccnt(0);
  }
#else
  for (int i = tid; i < 16 * DIM; i += 256) As[i] = Qn[(size_t)m0 * DIM + i];
#endif
  __syncthreads();

  int r   = lane & 15;
  int kbA = (lane >> 4) * 16;
  int kbB = (lane >> 4) * 16;
  int rb  = (lane >> 4) * 8;

  float ts[KTOP]; int ti[KTOP];
#pragma unroll
  for (int j = 0; j < KTOP; ++j) { ts[j] = -INFINITY; ti[j] = 0; }

  int cbase = chunk * CHUNK;
  for (int it = 0; it < 8; ++it) {              // 8 * 256 = 2048 >= CHUNK
    int n0 = cbase + it * 256 + wave * 32;      // wave-uniform
    bool v0ok = (n0 + 16 <= cbase + CHUNK);     // tile 0 valid (uniform)
    bool v1ok = (n0 + 32 <= cbase + CHUNK);     // tile 1 valid (uniform)
    v8f c0 = {}, c1 = {};
    const unsigned short* brow0 = Bn + (size_t)(n0 + r) * DIM;
    const unsigned short* brow1 = Bn + (size_t)(n0 + 16 + r) * DIM;
    if (v1ok) {                                 // EXEC all-1s inside: WMMA-legal
      __builtin_prefetch(brow0 + DIM * 256, 0, 0);
#pragma unroll 4
      for (int k0 = 0; k0 < DIM; k0 += 32) {
        Frag a, b0, b1;
        a.q[0]  = *(const u32x4*)&As[r * DIM + k0 + kbA];
        a.q[1]  = *(const u32x4*)&As[r * DIM + k0 + kbA + 8];
        b0.q[0] = *(const u32x4*)&brow0[k0 + kbB];
        b0.q[1] = *(const u32x4*)&brow0[k0 + kbB + 8];
        b1.q[0] = *(const u32x4*)&brow1[k0 + kbB];
        b1.q[1] = *(const u32x4*)&brow1[k0 + kbB + 8];
        c0 = wmma_bf16(a.v, b0.v, c0);
        c1 = wmma_bf16(a.v, b1.v, c1);
      }
    } else if (v0ok) {
#pragma unroll 4
      for (int k0 = 0; k0 < DIM; k0 += 32) {
        Frag a, b0;
        a.q[0]  = *(const u32x4*)&As[r * DIM + k0 + kbA];
        a.q[1]  = *(const u32x4*)&As[r * DIM + k0 + kbA + 8];
        b0.q[0] = *(const u32x4*)&brow0[k0 + kbB];
        b0.q[1] = *(const u32x4*)&brow0[k0 + kbB + 8];
        c0 = wmma_bf16(a.v, b0.v, c0);
      }
    }
#pragma unroll
    for (int j = 0; j < 8; ++j) {
      int row = m0 + rb + j;
      float s0 = -INFINITY, s1 = -INFINITY;
      if (v0ok) s0 = c0[j] * mask[(size_t)row * NANS + n0 + r];
      if (v1ok) s1 = c1[j] * mask[(size_t)row * NANS + n0 + 16 + r];
      Sc[wave * 512 +       (rb + j) * 16 + r] = s0;
      Sc[wave * 512 + 256 + (rb + j) * 16 + r] = s1;
    }
    __syncthreads();
    if (tid < 16) {                              // per-row serial top-10 insertion
      int base_n = cbase + it * 256;
      for (int w = 0; w < 8; ++w) {
        for (int t = 0; t < 2; ++t) {
#pragma unroll 1
          for (int cc = 0; cc < 16; ++cc) {
            float s = Sc[w * 512 + t * 256 + tid * 16 + cc];
            if (s > ts[KTOP - 1]) {
              int idx = base_n + w * 32 + t * 16 + cc;
              int j = KTOP - 1;
              while (j > 0 && ts[j - 1] < s) { ts[j] = ts[j - 1]; ti[j] = ti[j - 1]; --j; }
              ts[j] = s; ti[j] = idx;
            }
          }
        }
      }
    }
    __syncthreads();
  }
  if (tid < 16) {
    int row = m0 + tid;
#pragma unroll
    for (int j = 0; j < KTOP; ++j) {
      partS[((size_t)row * NCHUNKS + chunk) * KTOP + j] = ts[j];
      partI[((size_t)row * NCHUNKS + chunk) * KTOP + j] = ti[j];
    }
  }
}

// ---------------------------------------------------------------------------
// Merge 64 sorted-descending 10-lists per row into final top-10.
// ---------------------------------------------------------------------------
__global__ void merge_topk_kernel(const float* __restrict__ partS, const int* __restrict__ partI,
                                  float* __restrict__ outS, float* __restrict__ outIf,
                                  int* __restrict__ outIdx)
{
  int row = blockIdx.x * blockDim.x + threadIdx.x;
  if (row >= BATCH) return;
  float ts[KTOP]; int ti[KTOP];
#pragma unroll
  for (int j = 0; j < KTOP; ++j) { ts[j] = -INFINITY; ti[j] = 0; }
  const float* ps = partS + (size_t)row * NCHUNKS * KTOP;
  const int*   pi = partI + (size_t)row * NCHUNKS * KTOP;
  for (int c = 0; c < NCHUNKS; ++c) {
    for (int j = 0; j < KTOP; ++j) {
      float s = ps[c * KTOP + j];
      if (!(s > ts[KTOP - 1])) break;      // chunk lists sorted descending
      int idx = pi[c * KTOP + j];
      int p = KTOP - 1;
      while (p > 0 && ts[p - 1] < s) { ts[p] = ts[p - 1]; ti[p] = ti[p - 1]; --p; }
      ts[p] = s; ti[p] = idx;
    }
  }
#pragma unroll
  for (int j = 0; j < KTOP; ++j) {
    outS  [row * KTOP + j] = ts[j];
    outIf [row * KTOP + j] = (float)ti[j];
    outIdx[row * KTOP + j] = ti[j];
  }
}

// ---------------------------------------------------------------------------
// Gather top-k answer embeddings: one block per (row, k), copies 768 floats.
// ---------------------------------------------------------------------------
__global__ void gather_kernel(const int* __restrict__ outIdx, const float* __restrict__ emb,
                              float* __restrict__ outEmb)
{
  int rk = blockIdx.x;
  int idx = outIdx[rk];
  if (idx < 0 || idx >= NANS) idx = 0;
  const float* src = emb + (size_t)idx * DIM;
  float* dst = outEmb + (size_t)rk * DIM;
  for (int i = threadIdx.x; i < DIM; i += 256) dst[i] = src[i];
}

extern "C" void kernel_launch(void* const* d_in, const int* in_sizes, int n_in,
                              void* d_out, int out_size, void* d_ws, size_t ws_size,
                              hipStream_t stream)
{
  (void)in_sizes; (void)n_in; (void)out_size; (void)ws_size;
  const float* visual = (const float*)d_in[0];
  // d_in[1] (text), d_in[4..7] (wq,bq,wk,bk) are dead: softmax over 1 key == 1.
  const float* mask   = (const float*)d_in[2];
  const float* ansEmb = (const float*)d_in[3];
  const float* wv = (const float*)d_in[8];
  const float* bv = (const float*)d_in[9];
  const float* wo = (const float*)d_in[10];
  const float* bo = (const float*)d_in[11];
  const float* fw = (const float*)d_in[12];
  const float* fb = (const float*)d_in[13];
  const float* lg = (const float*)d_in[14];
  const float* lb = (const float*)d_in[15];
  const float* sw = (const float*)d_in[16];
  const float* sb = (const float*)d_in[17];

  char* ws = (char*)d_ws;
  size_t off = 0;
  unsigned short* Bn = (unsigned short*)(ws + off); off += (size_t)NANS * DIM * 2;
  float* v0   = (float*)(ws + off); off += (size_t)BATCH * DIM * 4;
  float* att  = (float*)(ws + off); off += (size_t)BATCH * DIM * 4;
  float* hbuf = (float*)(ws + off); off += (size_t)BATCH * DIM * 4;
  float* fus  = (float*)(ws + off); off += (size_t)BATCH * DIM * 4;
  float* fpr  = (float*)(ws + off); off += (size_t)BATCH * DIM * 4;
  unsigned short* Qn = (unsigned short*)(ws + off); off += (size_t)BATCH * DIM * 2;
  float* partS = (float*)(ws + off); off += (size_t)BATCH * NCHUNKS * KTOP * 4;
  int*   partI = (int*)(ws + off);   off += (size_t)BATCH * NCHUNKS * KTOP * 4;
  int*   oIdx  = (int*)(ws + off);

  float* outS   = (float*)d_out;
  float* outIf  = outS + BATCH * KTOP;
  float* outEmb = outIf + BATCH * KTOP;

  // Normalize 128000 answer rows -> bf16 (197 MB, ~fits global L2).
  l2norm_bf16_kernel<<<NANS / 8, 256, 0, stream>>>(ansEmb, Bn, NANS);

  dim3 gg(DIM / 16, BATCH / 16);
  // attended = (visual @ wv.T + bv) @ wo.T + bo
  gemm_bt_kernel<<<gg, 32, 0, stream>>>(visual, DIM, wv, DIM, bv, v0,  DIM, DIM, 0);
  gemm_bt_kernel<<<gg, 32, 0, stream>>>(v0,     DIM, wo, DIM, bo, att, DIM, DIM, 0);
  // h = visual @ fw[:, :768].T + fb  ;  h += attended @ fw[:, 768:].T
  gemm_bt_kernel<<<gg, 32, 0, stream>>>(visual, DIM, fw,       2 * DIM, fb,      hbuf, DIM, DIM, 0);
  gemm_bt_kernel<<<gg, 32, 0, stream>>>(att,    DIM, fw + DIM, 2 * DIM, nullptr, hbuf, DIM, DIM, 1);
  // LayerNorm + GELU
  ln_gelu_kernel<<<BATCH / 8, 256, 0, stream>>>(hbuf, lg, lb, fus);
  // fused_proj = fused @ sim_w.T + sim_b ; then L2-normalize -> bf16
  gemm_bt_kernel<<<gg, 32, 0, stream>>>(fus, DIM, sw, DIM, sb, fpr, DIM, DIM, 0);
  l2norm_bf16_kernel<<<BATCH / 8, 256, 0, stream>>>(fpr, Qn, BATCH);

  // 201-GFLOP similarity GEMM fused with mask + per-chunk top-10.
  sim_topk_kernel<<<dim3(MT, NCHUNKS), 256, 0, stream>>>(Qn, Bn, mask, partS, partI);
  // Merge partials, emit scores/indices, gather embeddings.
  merge_topk_kernel<<<(BATCH + 255) / 256, 256, 0, stream>>>(partS, partI, outS, outIf, oIdx);
  gather_kernel<<<BATCH * KTOP, 256, 0, stream>>>(oIdx, ansEmb, outEmb);
}
